// Autoformer_23837068493394
// MI455X (gfx1250) — matrix operand order repaired and checked by reading, not compiled, not run on hardware
//
#include <hip/hip_runtime.h>

// ---------------- problem constants ----------------
#define BB   1024      // batch
#define LE   96        // encoder length
#define LDd  116       // decoder length (96 + EXT)
#define DFi  5         // input features
#define DM   256       // d_model
#define DFF  1024      // d_ff
#define EXTN 20
#define TOPK_ENC 13    // int(3*ln(96))
#define TOPK_DEC 14    // int(3*ln(116))

// ---------------- WMMA fragment types ----------------
typedef __attribute__((ext_vector_type(16))) __bf16        v16bf;
typedef __attribute__((ext_vector_type(8)))  float         v8f;
typedef __attribute__((ext_vector_type(4)))  float         f32x4;
typedef __attribute__((ext_vector_type(4)))  unsigned int  u32x4;
typedef __attribute__((ext_vector_type(2)))  unsigned int  u32x2;

union ABFrag { v16bf v; unsigned short s[16]; unsigned int u[8]; u32x4 q[2]; };

__device__ __forceinline__ unsigned short f2bf(float f) {
  union { float f; unsigned int u; } c; c.f = f;
  unsigned int u = c.u;
  u += 0x7fffu + ((u >> 16) & 1u);   // round-to-nearest-even
  return (unsigned short)(u >> 16);
}

// Fragment element addresses in a bf16 row (ushort units), per ISA 7.12.2:
//  A 16x32 : lane m = lane%16, half = lane/16
//            q[0] at  row*K + kb + half*8        (K = 0..7  / 8..15)
//            q[1] at  row*K + kb + half*8  + 16  (K = 16..23/ 24..31)
//  B 32x16 : lane n = lane%16
//            q[0] at  row*K + kb + half*8        (K = 0..7  / 8..15)
//            q[1] at  row*K + kb + half*8  + 16  (K = 16..23/ 24..31)

// =====================================================================
// Projection GEMM (K = M = 256, bf16 in, f32 accumulate):
//   Y[b*outL + t, m] = sum_k X[b*inL + t, k] * W[m,k] + bias[m]
// one wave per 16x16 tile; Ntok multiple of 16; optional f32/bf16 outputs.
// =====================================================================
__global__ __launch_bounds__(256) void k_gemm_bias(
    const unsigned short* __restrict__ X, const unsigned short* __restrict__ W,
    const float* __restrict__ bias, float* __restrict__ Yf,
    unsigned short* __restrict__ Yh, int Ntok, int inL, int outL)
{
  int wave = blockIdx.x * 8 + (threadIdx.x >> 5);
  int tn = wave >> 4, tm = wave & 15;      // tilesM = 256/16 = 16
  int r0 = tn * 16;
  if (r0 >= Ntok) return;
  int m0   = tm * 16;
  int lane = threadIdx.x & 31;
  int lm = lane & 15, half = lane >> 4;

  const unsigned short* xrow = X + (size_t)(r0 + lm) * DM;
  const unsigned short* wrow = W + (size_t)(m0 + lm) * DM;

  v8f acc = {0.f,0.f,0.f,0.f,0.f,0.f,0.f,0.f};
  #pragma unroll
  for (int kb = 0; kb < DM; kb += 32) {
    ABFrag a, b;
    a.q[0] = *(const u32x4*)(xrow + kb + half * 8);
    a.q[1] = *(const u32x4*)(xrow + kb + half * 8 + 16);
    b.q[0] = *(const u32x4*)(wrow + kb + half * 8);
    b.q[1] = *(const u32x4*)(wrow + kb + half * 8 + 16);
    acc = __builtin_amdgcn_wmma_f32_16x16x32_bf16(false, a.v, false, b.v,
                                                  (short)0, acc, false, false);
  }
  float bcol = bias[m0 + lm];
  int outBase = r0;
  if (inL != outL) {                 // tiles never straddle batches (96 % 16 == 0)
    int bch = r0 / inL;
    outBase = bch * outL + (r0 - bch * inL);
  }
  #pragma unroll
  for (int r = 0; r < 8; ++r) {
    int orow = outBase + r + 8 * half;
    size_t o = (size_t)orow * DM + m0 + lm;
    float val = acc[r] + bcol;
    if (Yf) Yf[o] = val;
    if (Yh) Yh[o] = f2bf(val);
  }
}

// =====================================================================
// Per-batch Gram: G[b,i,j] = sum_c Q[b,i,c] * K[b,j,c]   (i,j < L), bf16 in
// =====================================================================
__global__ __launch_bounds__(256) void k_gram(
    const unsigned short* __restrict__ Q, const unsigned short* __restrict__ Kc,
    float* __restrict__ G, int L)
{
  int tiles = (L + 15) >> 4;
  int wpb   = tiles * tiles;
  int wave  = blockIdx.x * 8 + (threadIdx.x >> 5);
  int b = wave / wpb;
  if (b >= BB) return;
  int w  = wave % wpb;
  int ti = w / tiles, tj = w % tiles;
  int lane = threadIdx.x & 31;
  int lm = lane & 15, half = lane >> 4;

  int ri = ti * 16 + lm;  bool iok = ri < L;
  int rj = tj * 16 + lm;  bool jok = rj < L;
  const unsigned short* qrow = Q  + ((size_t)b * L + (iok ? ri : 0)) * DM;
  const unsigned short* krow = Kc + ((size_t)b * L + (jok ? rj : 0)) * DM;

  v8f acc = {0.f,0.f,0.f,0.f,0.f,0.f,0.f,0.f};
  #pragma unroll
  for (int kb = 0; kb < DM; kb += 32) {
    ABFrag a, bf;
    a.q[0]  = *(const u32x4*)(qrow + kb + half * 8);
    a.q[1]  = *(const u32x4*)(qrow + kb + half * 8 + 16);
    bf.q[0] = *(const u32x4*)(krow + kb + half * 8);
    bf.q[1] = *(const u32x4*)(krow + kb + half * 8 + 16);
    if (!iok) { a.q[0]  = (u32x4)(0u); a.q[1]  = (u32x4)(0u); }
    if (!jok) { bf.q[0] = (u32x4)(0u); bf.q[1] = (u32x4)(0u); }
    acc = __builtin_amdgcn_wmma_f32_16x16x32_bf16(false, a.v, false, bf.v,
                                                  (short)0, acc, false, false);
  }
  #pragma unroll
  for (int r = 0; r < 8; ++r) {
    int i = ti * 16 + r + 8 * half, j = tj * 16 + lm;
    if (i < L && j < L) G[((size_t)b * L + i) * L + j] = acc[r];
  }
}

// =====================================================================
// Fused FFN: Y = relu(X @ W1^T) @ W2^T  (bf16 in, hidden tile in LDS as bf16)
// block = 16 tokens, 8 waves.
// =====================================================================
__global__ __launch_bounds__(256) void k_ffn(
    const unsigned short* __restrict__ X, const unsigned short* __restrict__ W1,
    const unsigned short* __restrict__ W2, float* __restrict__ Y)
{
  __shared__ unsigned short hid[16 * DFF];   // 32 KB bf16
  int r0 = blockIdx.x * 16;
  int wv = threadIdx.x >> 5, lane = threadIdx.x & 31;
  int lm = lane & 15, half = lane >> 4;
  const unsigned short* xrow = X + (size_t)(r0 + lm) * DM;

  // phase 1: wave wv -> hidden columns [wv*128, wv*128+128)
  for (int hc = wv * 128; hc < wv * 128 + 128; hc += 16) {
    const unsigned short* wrow = W1 + (size_t)(hc + lm) * DM;
    v8f acc = {0.f,0.f,0.f,0.f,0.f,0.f,0.f,0.f};
    #pragma unroll
    for (int kb = 0; kb < DM; kb += 32) {
      ABFrag a, b;
      a.q[0] = *(const u32x4*)(xrow + kb + half * 8);
      a.q[1] = *(const u32x4*)(xrow + kb + half * 8 + 16);
      b.q[0] = *(const u32x4*)(wrow + kb + half * 8);
      b.q[1] = *(const u32x4*)(wrow + kb + half * 8 + 16);
      acc = __builtin_amdgcn_wmma_f32_16x16x32_bf16(false, a.v, false, b.v,
                                                    (short)0, acc, false, false);
    }
    #pragma unroll
    for (int r = 0; r < 8; ++r) {
      float vr = acc[r] > 0.f ? acc[r] : 0.f;         // ReLU
      hid[(r + 8 * half) * DFF + hc + lm] = f2bf(vr);
    }
  }
  __syncthreads();

  // phase 2: wave wv -> output column tiles 2wv, 2wv+1 ; A from LDS (b128)
  for (int mt = wv * 2; mt < wv * 2 + 2; ++mt) {
    int m0 = mt * 16;
    const unsigned short* w2row = W2 + (size_t)(m0 + lm) * DFF;
    const unsigned short* hrow  = hid + lm * DFF;
    v8f acc = {0.f,0.f,0.f,0.f,0.f,0.f,0.f,0.f};
    #pragma unroll 4
    for (int kb = 0; kb < DFF; kb += 32) {
      ABFrag a, b;
      a.q[0] = *(const u32x4*)(hrow + kb + half * 8);
      a.q[1] = *(const u32x4*)(hrow + kb + half * 8 + 16);
      b.q[0] = *(const u32x4*)(w2row + kb + half * 8);
      b.q[1] = *(const u32x4*)(w2row + kb + half * 8 + 16);
      acc = __builtin_amdgcn_wmma_f32_16x16x32_bf16(false, a.v, false, b.v,
                                                    (short)0, acc, false, false);
    }
    #pragma unroll
    for (int r = 0; r < 8; ++r) {
      int row = r0 + r + 8 * half;
      Y[(size_t)row * DM + m0 + lm] = acc[r];
    }
  }
}

// =====================================================================
// fp32 -> bf16 tensor conversion (vectorized, n multiple of 4)
// =====================================================================
__global__ void k_cvt_bf16(const float* __restrict__ in,
                           unsigned short* __restrict__ out, size_t n4)
{
  size_t i = (size_t)blockIdx.x * blockDim.x + threadIdx.x;
  size_t stride = (size_t)gridDim.x * blockDim.x;
  for (; i < n4; i += stride) {
    f32x4 v = *(const f32x4*)(in + 4 * i);
    u32x2 p;
    p[0] = (unsigned)f2bf(v[0]) | ((unsigned)f2bf(v[1]) << 16);
    p[1] = (unsigned)f2bf(v[2]) | ((unsigned)f2bf(v[3]) << 16);
    *(u32x2*)(out + 4 * i) = p;
  }
}

// =====================================================================
// scalar helper kernels
// =====================================================================
__global__ void k_add(const float* __restrict__ a, const float* __restrict__ b,
                      float* __restrict__ z, size_t n)
{
  size_t i = (size_t)blockIdx.x * blockDim.x + threadIdx.x;
  size_t stride = (size_t)gridDim.x * blockDim.x;
  for (; i < n; i += stride) z[i] = a[i] + b[i];
}

// series_decomp: s = x - movmean25(x) (replicate pad); optional tacc += movmean
__global__ __launch_bounds__(256) void k_decomp(
    const float* __restrict__ x, float* __restrict__ s,
    float* __restrict__ tacc, int L)
{
  int b = blockIdx.x / L, t = blockIdx.x % L;
  int c = threadIdx.x;
  const float* xb = x + (size_t)b * L * DM;
  float acc = 0.f;
  #pragma unroll
  for (int d = -12; d <= 12; ++d) {
    int tt = t + d; tt = tt < 0 ? 0 : (tt >= L ? L - 1 : tt);
    acc += xb[(size_t)tt * DM + c];
  }
  float m = acc * (1.f / 25.f);
  size_t idx = ((size_t)b * L + t) * DM + c;
  s[idx] = x[idx] - m;
  if (tacc) tacc[idx] += m;
}

// circular conv3, Cin=5 -> Cout=256
__global__ __launch_bounds__(256) void k_conv5to256(
    const float* __restrict__ x, const float* __restrict__ W,
    float* __restrict__ out, int L)
{
  int b = blockIdx.x / L, t = blockIdx.x % L;
  int tid = threadIdx.x;
  __shared__ float win[15];
  if (tid < 15) {
    int tau = tid / 5, c = tid % 5;
    int tt = (t + tau - 1 + L) % L;
    win[tau * 5 + c] = x[((size_t)b * L + tt) * DFi + c];
  }
  __syncthreads();
  const float* w = W + tid * 15;       // W[o][c][tau]
  float acc = 0.f;
  #pragma unroll
  for (int c = 0; c < 5; ++c)
    #pragma unroll
    for (int tau = 0; tau < 3; ++tau)
      acc += win[tau * 5 + c] * w[c * 3 + tau];
  out[((size_t)b * L + t) * DM + tid] = acc;
}

// circular conv3, Cin=256 -> Cout=5, accumulated into trend
__global__ __launch_bounds__(256) void k_conv_trend(
    const float* __restrict__ x, const float* __restrict__ W,
    float* __restrict__ trend)
{
  int b = blockIdx.x / LDd, t = blockIdx.x % LDd;
  int c = threadIdx.x;
  __shared__ float red[5][256];
  float xm = x[((size_t)b * LDd + (t - 1 + LDd) % LDd) * DM + c];
  float x0 = x[((size_t)b * LDd + t) * DM + c];
  float xp = x[((size_t)b * LDd + (t + 1) % LDd) * DM + c];
  #pragma unroll
  for (int o = 0; o < 5; ++o) {
    const float* w = W + ((size_t)o * DM + c) * 3;
    red[o][c] = xm * w[0] + x0 * w[1] + xp * w[2];
  }
  __syncthreads();
  for (int sft = 128; sft > 0; sft >>= 1) {
    if (c < sft)
      #pragma unroll
      for (int o = 0; o < 5; ++o) red[o][c] += red[o][c + sft];
    __syncthreads();
  }
  if (c < 5) trend[((size_t)b * LDd + t) * 5 + c] += red[c][0];
}

// mean_value[b,tau] = (1/256) * sum_t G[b, (t+tau)%L, t]
__global__ void k_corr_mean(const float* __restrict__ G,
                            float* __restrict__ mv, int L)
{
  int b = blockIdx.x, tau = threadIdx.x;
  if (tau >= L) return;
  const float* g = G + (size_t)b * L * L;
  float acc = 0.f;
  for (int t = 0; t < L; ++t) acc += g[(size_t)((t + tau) % L) * L + t];
  mv[b * L + tau] = acc * (1.f / 256.f);
}

// per-batch top-k + softmax (k <= 14, L <= 116)
__global__ void k_topk(const float* __restrict__ mv, float* __restrict__ wts,
                       int* __restrict__ dly, int L, int K)
{
  __shared__ float vals[128];
  int b = blockIdx.x, tid = threadIdx.x;
  vals[tid] = (tid < L) ? mv[b * L + tid] : -3e38f;
  __syncthreads();
  if (tid == 0) {
    float tv[16]; int ti[16];
    for (int j = 0; j < K; ++j) {
      int bi = 0; float bv = -3e38f;
      for (int t = 0; t < L; ++t)
        if (vals[t] > bv) { bv = vals[t]; bi = t; }
      tv[j] = bv; ti[j] = bi; vals[bi] = -3e38f;
    }
    float mx = tv[0], se = 0.f;
    for (int j = 0; j < K; ++j) { tv[j] = __expf(tv[j] - mx); se += tv[j]; }
    float inv = 1.f / se;
    for (int j = 0; j < K; ++j) { wts[b * 16 + j] = tv[j] * inv; dly[b * 16 + j] = ti[j]; }
  }
}

// delay aggregation: out[b,t,c] = sum_j w[b,j] * V[b,(t+d[b,j])%L, c]
__global__ __launch_bounds__(256) void k_agg(
    const float* __restrict__ V, const float* __restrict__ wts,
    const int* __restrict__ dly, float* __restrict__ out, int L, int K)
{
  int b = blockIdx.x / L, t = blockIdx.x % L;
  int c = threadIdx.x;
  const float* vb = V + (size_t)b * L * DM;
  float acc = 0.f;
  for (int j = 0; j < K; ++j) {
    float w = wts[b * 16 + j];
    int   d = dly[b * 16 + j];
    acc += w * vb[(size_t)((t + d) % L) * DM + c];
  }
  out[((size_t)b * L + t) * DM + c] = acc;
}

// layernorm part 1: xh = (x-mu)/sqrt(var+1e-5)*g + beta   (per token)
__global__ __launch_bounds__(256) void k_ln(
    const float* __restrict__ x, const float* __restrict__ g,
    const float* __restrict__ be, float* __restrict__ out, int L)
{
  int b = blockIdx.x / L, t = blockIdx.x % L;
  int c = threadIdx.x;
  __shared__ float red[256];
  size_t idx = ((size_t)b * L + t) * DM + c;
  float v = x[idx];
  red[c] = v; __syncthreads();
  for (int s = 128; s > 0; s >>= 1) { if (c < s) red[c] += red[c + s]; __syncthreads(); }
  float mu = red[0] * (1.f / DM);
  __syncthreads();
  float dv = v - mu;
  red[c] = dv * dv; __syncthreads();
  for (int s = 128; s > 0; s >>= 1) { if (c < s) red[c] += red[c + s]; __syncthreads(); }
  float var = red[0] * (1.f / DM);
  out[idx] = dv * rsqrtf(var + 1e-5f) * g[c] + be[c];
}

// layernorm part 2: subtract per-(b,c) mean over L
__global__ __launch_bounds__(256) void k_subcol(
    const float* __restrict__ xh, float* __restrict__ out, int L)
{
  int b = blockIdx.x, c = threadIdx.x;
  float s = 0.f;
  for (int t = 0; t < L; ++t) s += xh[((size_t)b * L + t) * DM + c];
  float mu = s / (float)L;
  for (int t = 0; t < L; ++t) {
    size_t i = ((size_t)b * L + t) * DM + c;
    out[i] = xh[i] - mu;
  }
}

// initial decomposition: seasonal_init (front-padded), trend_init (mean-tailed)
__global__ void k_init(const float* __restrict__ x,
                       float* __restrict__ seasonal, float* __restrict__ trend)
{
  int b = blockIdx.x, tid = threadIdx.x;
  __shared__ float meanb[5];
  if (tid < 5) {
    float s = 0.f;
    for (int t = 0; t < LE; ++t) s += x[((size_t)b * LE + t) * DFi + tid];
    meanb[tid] = s / (float)LE;
  }
  __syncthreads();
  for (int i = tid; i < LE * 5; i += blockDim.x) {
    int t = i / 5, c = i % 5;
    float acc = 0.f;
    for (int d = -12; d <= 12; ++d) {
      int tt = t + d; tt = tt < 0 ? 0 : (tt > LE - 1 ? LE - 1 : tt);
      acc += x[((size_t)b * LE + tt) * DFi + c];
    }
    float m = acc * (1.f / 25.f);
    trend[((size_t)b * LDd + t) * 5 + c] = m;
    seasonal[((size_t)b * LDd + (t + EXTN)) * 5 + c] =
        x[((size_t)b * LE + t) * DFi + c] - m;
  }
  for (int i = tid; i < EXTN * 5; i += blockDim.x) {
    int j = i / 5, c = i % 5;
    trend[((size_t)b * LDd + LE + j) * 5 + c] = meanb[c];
    seasonal[((size_t)b * LDd + j) * 5 + c]  = 0.f;
  }
}

// final: project last dec row, add trend, batchnorm, fc -> d_out[b]
__global__ __launch_bounds__(256) void k_final(
    const float* __restrict__ dec, const float* __restrict__ trend,
    const float* __restrict__ projW, const float* __restrict__ projb,
    const float* __restrict__ bng, const float* __restrict__ bnb,
    const float* __restrict__ bnrm, const float* __restrict__ bnrv,
    const float* __restrict__ fcW, const float* __restrict__ fcb,
    float* __restrict__ out)
{
  int b = blockIdx.x, c = threadIdx.x;
  __shared__ float red[5][256];
  float xv = dec[((size_t)b * LDd + (LDd - 1)) * DM + c];
  #pragma unroll
  for (int o = 0; o < 5; ++o) red[o][c] = xv * projW[o * DM + c];
  __syncthreads();
  for (int s = 128; s > 0; s >>= 1) {
    if (c < s)
      #pragma unroll
      for (int o = 0; o < 5; ++o) red[o][c] += red[o][c + s];
    __syncthreads();
  }
  if (c == 0) {
    float acc = 0.f;
    for (int j = 0; j < EXTN; ++j) {
      int o = j % 5;
      float last = trend[((size_t)b * LDd + (LDd - 1)) * 5 + o] + red[o][0] + projb[o];
      float h = (last - bnrm[j]) * rsqrtf(bnrv[j] + 1e-5f) * bng[j] + bnb[j];
      acc += h * fcW[j];
    }
    out[b] = acc + fcb[0];
  }
}

// =====================================================================
// host orchestration
// =====================================================================
extern "C" void kernel_launch(void* const* d_in, const int* in_sizes, int n_in,
                              void* d_out, int out_size, void* d_ws, size_t ws_size,
                              hipStream_t stream) {
  const float* x       = (const float*)d_in[0];
  const float* we_enc  = (const float*)d_in[1];
  const float* we_dec  = (const float*)d_in[2];
  const float* enc_Wq  = (const float*)d_in[3];
  const float* enc_bq  = (const float*)d_in[4];
  const float* enc_Wk  = (const float*)d_in[5];
  const float* enc_bk  = (const float*)d_in[6];
  const float* enc_Wv  = (const float*)d_in[7];
  const float* enc_bv  = (const float*)d_in[8];
  const float* enc_Wo  = (const float*)d_in[9];
  const float* enc_bo  = (const float*)d_in[10];
  const float* enc_W1  = (const float*)d_in[11];
  const float* enc_W2  = (const float*)d_in[12];
  const float* enc_lng = (const float*)d_in[13];
  const float* enc_lnb = (const float*)d_in[14];
  const float* dsWq = (const float*)d_in[15]; const float* dsbq = (const float*)d_in[16];
  const float* dsWk = (const float*)d_in[17]; const float* dsbk = (const float*)d_in[18];
  const float* dsWv = (const float*)d_in[19]; const float* dsbv = (const float*)d_in[20];
  const float* dsWo = (const float*)d_in[21]; const float* dsbo = (const float*)d_in[22];
  const float* dcWq = (const float*)d_in[23]; const float* dcbq = (const float*)d_in[24];
  const float* dcWk = (const float*)d_in[25]; const float* dcbk = (const float*)d_in[26];
  const float* dcWv = (const float*)d_in[27]; const float* dcbv = (const float*)d_in[28];
  const float* dcWo = (const float*)d_in[29]; const float* dcbo = (const float*)d_in[30];
  const float* dec_W1  = (const float*)d_in[31];
  const float* dec_W2  = (const float*)d_in[32];
  const float* dec_Wt  = (const float*)d_in[33];
  const float* dec_lng = (const float*)d_in[34];
  const float* dec_lnb = (const float*)d_in[35];
  const float* proj_W  = (const float*)d_in[36];
  const float* proj_b  = (const float*)d_in[37];
  const float* bn_g  = (const float*)d_in[38];
  const float* bn_b  = (const float*)d_in[39];
  const float* bn_rm = (const float*)d_in[40];
  const float* bn_rv = (const float*)d_in[41];
  const float* fc_W  = (const float*)d_in[42];
  const float* fc_b  = (const float*)d_in[43];

  // ---------- workspace layout ----------
  const size_t nE = (size_t)BB * LE  * DM;   // enc activations
  const size_t nD = (size_t)BB * LDd * DM;   // dec-sized activations
  const size_t nG = (size_t)BB * LDd * LDd;  // Gram
  float* ws    = (float*)d_ws;
  float* bufE  = ws;                // fp32 region
  float* bufD  = bufE + nE;
  float* buf1  = bufD + nD;
  float* buf2  = buf1 + nD;
  float* buf3  = buf2 + nD;
  float* buf4  = buf3 + nD;         // t1+t2+t3 accumulator
  float* bufG  = buf4 + nD;
  float* mv    = bufG + nG;                         // B*LDd
  float* wts   = mv + (size_t)BB * LDd;             // B*16
  int*   dly   = (int*)(wts + (size_t)BB * 16);     // B*16
  float* seas  = (float*)(dly + (size_t)BB * 16);   // B*LDd*5
  float* trend = seas + (size_t)BB * LDd * 5;       // B*LDd*5

  // bf16 region (ushort), 16B aligned (all prior counts are multiples of 4)
  unsigned short* u16 = (unsigned short*)(trend + (size_t)BB * LDd * 5);
  unsigned short* bfX  = u16;            // bf16 of gemm input X
  unsigned short* bfKV = bfX  + nD;      // bf16 of cross-attn KV source (enc)
  unsigned short* bfQ  = bfKV + nE;      // Q (bf16 only)
  unsigned short* bfK  = bfQ  + nD;      // K (bf16 only)
  unsigned short* bfV  = bfK  + nD;      // bf16 of Vagg / FFN input
  unsigned short* wb   = bfV  + nD;      // converted weights
  const size_t WP = (size_t)2 * DM * DM;     // proj weight tensor (both layers)
  const size_t WF = (size_t)2 * DFF * DM;    // ffn weight tensor (both layers)
  unsigned short* w_eq = wb;            unsigned short* w_ek = w_eq + WP;
  unsigned short* w_ev = w_ek + WP;     unsigned short* w_eo = w_ev + WP;
  unsigned short* w_e1 = w_eo + WP;     unsigned short* w_e2 = w_e1 + WF;
  unsigned short* w_sq = w_e2 + WF;     unsigned short* w_sk = w_sq + WP;
  unsigned short* w_sv = w_sk + WP;     unsigned short* w_so = w_sv + WP;
  unsigned short* w_cq = w_so + WP;     unsigned short* w_ck = w_cq + WP;
  unsigned short* w_cv = w_ck + WP;     unsigned short* w_co = w_cv + WP;
  unsigned short* w_d1 = w_co + WP;     unsigned short* w_d2 = w_d1 + WF;

  auto cvt = [&](const float* src, unsigned short* dst, size_t n) {
    k_cvt_bf16<<<2048, 256, 0, stream>>>(src, dst, n / 4);
  };
  auto gemm = [&](const unsigned short* X, const unsigned short* W, const float* bias,
                  float* Yf, unsigned short* Yh, int Ntok, int inL, int outL) {
    int waves  = (Ntok / 16) * 16;
    k_gemm_bias<<<(waves + 7) / 8, 256, 0, stream>>>(X, W, bias, Yf, Yh, Ntok, inL, outL);
  };

  // ---------- one-time (per call) weight conversion ----------
  cvt(enc_Wq, w_eq, WP); cvt(enc_Wk, w_ek, WP); cvt(enc_Wv, w_ev, WP); cvt(enc_Wo, w_eo, WP);
  cvt(enc_W1, w_e1, WF); cvt(enc_W2, w_e2, WF);
  cvt(dsWq,   w_sq, WP); cvt(dsWk,   w_sk, WP); cvt(dsWv,   w_sv, WP); cvt(dsWo,   w_so, WP);
  cvt(dcWq,   w_cq, WP); cvt(dcWk,   w_ck, WP); cvt(dcWv,   w_cv, WP); cvt(dcWo,   w_co, WP);
  cvt(dec_W1, w_d1, WF); cvt(dec_W2, w_d2, WF);

  // autocorrelation attention layer: result left in buf2
  auto ac = [&](const float* Xq, int Lq, const float* Xkv, int Ls,
                const unsigned short* Wq, const float* bq,
                const unsigned short* Wk, const float* bk,
                const unsigned short* Wv, const float* bv,
                const unsigned short* Wo, const float* bo, int topK) {
    size_t nQ = (size_t)BB * Lq * DM;
    cvt(Xq, bfX, nQ);
    const unsigned short* kvsrc = bfX;
    if (Ls != Lq) {   // cross-attn: zero-pad projected K/V rows >= Ls
      cvt(Xkv, bfKV, (size_t)BB * Ls * DM);
      kvsrc = bfKV;
      hipMemsetAsync(bfK,  0, nQ * sizeof(unsigned short), stream);
      hipMemsetAsync(buf3, 0, nQ * sizeof(float), stream);
    }
    gemm(bfX,   Wq, bq, nullptr, bfQ,  BB * Lq, Lq, Lq);   // Q (bf16)
    gemm(kvsrc, Wk, bk, nullptr, bfK,  BB * Ls, Ls, Lq);   // K (bf16)
    gemm(kvsrc, Wv, bv, buf3,  nullptr, BB * Ls, Ls, Lq);  // V (fp32)
    int tiles = (Lq + 15) / 16;
    int waves = BB * tiles * tiles;
    k_gram<<<(waves + 7) / 8, 256, 0, stream>>>(bfQ, bfK, bufG, Lq);
    k_corr_mean<<<BB, 128, 0, stream>>>(bufG, mv, Lq);
    k_topk<<<BB, 128, 0, stream>>>(mv, wts, dly, Lq, topK);
    k_agg<<<BB * Lq, 256, 0, stream>>>(buf3, wts, dly, buf1, Lq, topK); // Vagg
    cvt(buf1, bfV, nQ);
    gemm(bfV, Wo, bo, buf2, nullptr, BB * Lq, Lq, Lq);     // attention out
  };

  const size_t WPl = (size_t)DM * DM, B256 = DM, WFl = (size_t)DFF * DM;

  // ---------- initial decomposition + embeddings ----------
  k_init<<<BB, 128, 0, stream>>>(x, seas, trend);
  k_conv5to256<<<BB * LE, 256, 0, stream>>>(x, we_enc, bufE, LE);

  // ---------- encoder ----------
  for (int i = 0; i < 2; ++i) {
    ac(bufE, LE, bufE, LE,
       w_eq + i * WPl, enc_bq + i * B256, w_ek + i * WPl, enc_bk + i * B256,
       w_ev + i * WPl, enc_bv + i * B256, w_eo + i * WPl, enc_bo + i * B256,
       TOPK_ENC);
    k_add<<<4096, 256, 0, stream>>>(bufE, buf2, buf3, nE);
    k_decomp<<<BB * LE, 256, 0, stream>>>(buf3, buf1, (float*)nullptr, LE);   // h
    cvt(buf1, bfV, nE);
    k_ffn<<<(BB * LE) / 16, 256, 0, stream>>>(bfV, w_e1 + i * WFl, w_e2 + i * WFl, buf2);
    k_add<<<4096, 256, 0, stream>>>(buf1, buf2, buf3, nE);
    k_decomp<<<BB * LE, 256, 0, stream>>>(buf3, bufE, (float*)nullptr, LE);
  }
  k_ln<<<BB * LE, 256, 0, stream>>>(bufE, enc_lng, enc_lnb, buf1, LE);
  k_subcol<<<BB, 256, 0, stream>>>(buf1, bufE, LE);

  // ---------- decoder ----------
  k_conv5to256<<<BB * LDd, 256, 0, stream>>>(seas, we_dec, bufD, LDd);
  for (int i = 0; i < 2; ++i) {
    hipMemsetAsync(buf4, 0, nD * sizeof(float), stream);
    // self attention
    ac(bufD, LDd, bufD, LDd,
       w_sq + i * WPl, dsbq + i * B256, w_sk + i * WPl, dsbk + i * B256,
       w_sv + i * WPl, dsbv + i * B256, w_so + i * WPl, dsbo + i * B256,
       TOPK_DEC);
    k_add<<<4096, 256, 0, stream>>>(bufD, buf2, buf3, nD);
    k_decomp<<<BB * LDd, 256, 0, stream>>>(buf3, bufD, buf4, LDd);       // t1
    // cross attention (enc K/V, zero-padded 96 -> 116)
    ac(bufD, LDd, bufE, LE,
       w_cq + i * WPl, dcbq + i * B256, w_ck + i * WPl, dcbk + i * B256,
       w_cv + i * WPl, dcbv + i * B256, w_co + i * WPl, dcbo + i * B256,
       TOPK_DEC);
    k_add<<<4096, 256, 0, stream>>>(bufD, buf2, buf3, nD);
    k_decomp<<<BB * LDd, 256, 0, stream>>>(buf3, bufD, buf4, LDd);       // t2
    // FFN
    cvt(bufD, bfV, nD);
    k_ffn<<<(BB * LDd) / 16, 256, 0, stream>>>(bfV, w_d1 + i * WFl, w_d2 + i * WFl, buf2);
    k_add<<<4096, 256, 0, stream>>>(bufD, buf2, buf3, nD);
    k_decomp<<<BB * LDd, 256, 0, stream>>>(buf3, bufD, buf4, LDd);       // t3
    // trend += circ_conv3(t1+t2+t3, Wt)
    k_conv_trend<<<BB * LDd, 256, 0, stream>>>(buf4, dec_Wt + (size_t)i * 5 * DM * 3, trend);
  }
  k_ln<<<BB * LDd, 256, 0, stream>>>(bufD, dec_lng, dec_lnb, buf1, LDd);
  k_subcol<<<BB, 256, 0, stream>>>(buf1, bufD, LDd);

  // ---------- final projection / batchnorm / fc ----------
  k_final<<<BB, 256, 0, stream>>>(bufD, trend, proj_W, proj_b,
                                  bn_g, bn_b, bn_rm, bn_rv, fc_W, fc_b,
                                  (float*)d_out);
}
